// StackLSTM_58282706207122
// MI455X (gfx1250) — compile-verified, matrix operand order checked by reading, not run on hardware
//
#include <hip/hip_runtime.h>
#include <hip/hip_bf16.h>
#include <math.h>

#define Tt   512
#define Bb   512
#define Ii   128
#define Hh   128
#define STK1 65      // stack slots (STACK+1)
#define GP   516     // padded gate row to break LDS bank conflicts

typedef __attribute__((ext_vector_type(16))) __bf16 v16bf;
typedef __attribute__((ext_vector_type(8)))  float  v8f;

union Acc { v8f v; float f[8]; };

__device__ __forceinline__ float sigf(float x) { return 1.0f / (1.0f + __expf(-x)); }

// Zero both stacks; slot 0 gets the learned initial states broadcast over batch.
__global__ void stack_init(float* __restrict__ hs, float* __restrict__ cs,
                           const float* __restrict__ ih, const float* __restrict__ ic) {
  int i = blockIdx.x * blockDim.x + threadIdx.x;
  const int n = STK1 * Bb * Hh;
  if (i >= n) return;
  const int h = i & (Hh - 1);
  float vh = 0.f, vc = 0.f;
  if (i < Bb * Hh) { vh = ih[h]; vc = ic[h]; }
  hs[i] = vh;
  cs[i] = vc;
}

// One workgroup owns 16 batch rows for all 511 steps. 16 waves; each wave owns
// two 16-wide gate column tiles with its bf16 weight fragments pinned in VGPRs.
__launch_bounds__(512)
__global__ void stacklstm(const float* __restrict__ xall,  // (T,B,I)
                          const int*   __restrict__ ops,   // (T,B)
                          const float* __restrict__ Wih,   // (4H,I)
                          const float* __restrict__ Whh,   // (4H,H)
                          const float* __restrict__ bih,
                          const float* __restrict__ bhh,
                          float* __restrict__ hs,          // (STK1,B,H) stack
                          float* __restrict__ cs,          // (STK1,B,H) stack
                          float* __restrict__ out)         // (T-1,B,H)
{
  __shared__ __align__(32) __bf16 lxa[4][32][16];  // x  A-fragments
  __shared__ __align__(32) __bf16 lha[4][32][16];  // h  A-fragments
  __shared__ float lg[16][GP];                     // gates f32

  const int tid  = threadIdx.x;
  const int lane = tid & 31;
  const int wid  = tid >> 5;            // wave 0..15
  const int b0   = blockIdx.x << 4;     // batch base row
  const int nlo  = lane & 15;
  const int g8   = (lane >> 4) << 3;    // +8 K-offset for upper lane half

  // ---- prologue: pin bf16 weight B-fragments + fused bias in registers ----
  v16bf wih[2][4], whh[2][4];
  float biasn[2];
#pragma unroll
  for (int nt = 0; nt < 2; ++nt) {
    const int ng = ((wid << 1) + nt) * 16 + nlo;   // gate column 0..511
    const float* rih = Wih + ng * Ii;
    const float* rhh = Whh + ng * Hh;
#pragma unroll
    for (int kk = 0; kk < 4; ++kk)
#pragma unroll
      for (int e = 0; e < 16; ++e) {
        const int v = e >> 1, hi = e & 1;
        const int k = (kk << 5) + ((v < 4) ? (v << 1) : (16 + ((v - 4) << 1))) + g8 + hi;
        wih[nt][kk][e] = (__bf16)rih[k];
        whh[nt][kk][e] = (__bf16)rhh[k];
      }
    biasn[nt] = bih[ng] + bhh[ng];
  }

  // staging decomposition: 2048 bf16 elems per matrix, 4 per thread, k-contiguous
  const int skk   = tid >> 7;                 // K-block 0..3
  const int slane = (tid >> 2) & 31;          // target lane slot
  const int sm    = slane & 15;               // batch row in tile
  const int sg8   = (slane >> 4) << 3;
  const int se0   = (tid & 3) << 2;           // element base 0,4,8,12
  const int sv    = se0 >> 1;                 // 0,2,4,6
  const int sk    = (skk << 5) + ((sv < 4) ? (sv << 1) : (16 + ((sv - 4) << 1))) + sg8;

  // elementwise decomposition: row em, 4 consecutive h columns
  const int em  = tid >> 5;
  const int eh0 = (tid & 31) << 2;

  // per-thread running stack pointers (replicated per row; no LDS, no barrier)
  int p_sm = 0;   // pointer for staging row sm
  int p_em = 0;   // pointer for elementwise row em

  for (int s = 0; s < Tt - 1; ++s) {
    // ---- pointer cumsum (registers, broadcast loads) ----
    p_sm += ops[s * Bb + b0 + sm];
    p_em += ops[s * Bb + b0 + em];

    // ---- phase 1: gather x_t and cur_h, convert bf16, store swizzled ----
    {
      const float4 xv = *(const float4*)(xall + (s * Bb + b0 + sm) * Ii + sk);
      __bf16* dx = &lxa[skk][slane][se0];
      dx[0] = (__bf16)xv.x; dx[1] = (__bf16)xv.y;
      dx[2] = (__bf16)xv.z; dx[3] = (__bf16)xv.w;
      const float4 hv = *(const float4*)(hs + (p_sm * Bb + b0 + sm) * Hh + sk);
      __bf16* dh = &lha[skk][slane][se0];
      dh[0] = (__bf16)hv.x; dh[1] = (__bf16)hv.y;
      dh[2] = (__bf16)hv.z; dh[3] = (__bf16)hv.w;
      // hide HBM latency of next step's input row behind WMMA + elementwise
      __builtin_prefetch(xall + ((s + 1) * Bb + b0 + sm) * Ii + sk, 0, 3);
    }
    __syncthreads();

    // ---- phase 2: gates = x@Wih^T + h@Whh^T + bias via bf16 WMMA ----
    {
      Acc a0, a1;
#pragma unroll
      for (int r = 0; r < 8; ++r) { a0.f[r] = biasn[0]; a1.f[r] = biasn[1]; }
#pragma unroll
      for (int kk = 0; kk < 4; ++kk) {
        const v16bf ax = *(const v16bf*)(&lxa[kk][lane][0]);
        const v16bf ah = *(const v16bf*)(&lha[kk][lane][0]);
        a0.v = __builtin_amdgcn_wmma_f32_16x16x32_bf16(false, ax, false, wih[0][kk],
                                                       (short)0, a0.v, false, false);
        a0.v = __builtin_amdgcn_wmma_f32_16x16x32_bf16(false, ah, false, whh[0][kk],
                                                       (short)0, a0.v, false, false);
        a1.v = __builtin_amdgcn_wmma_f32_16x16x32_bf16(false, ax, false, wih[1][kk],
                                                       (short)0, a1.v, false, false);
        a1.v = __builtin_amdgcn_wmma_f32_16x16x32_bf16(false, ah, false, whh[1][kk],
                                                       (short)0, a1.v, false, false);
      }
      const int n0 = (wid << 5) + nlo;
      const int mb = (lane >> 4) << 3;
#pragma unroll
      for (int r = 0; r < 8; ++r) {
        lg[mb + r][n0]      = a0.f[r];
        lg[mb + r][n0 + 16] = a1.f[r];
      }
    }
    __syncthreads();

    // ---- phase 3: LSTM cell, stack scatter, op-blend output ----
    {
      const int p      = p_em;
      const int op     = ops[(s + 1) * Bb + b0 + em];
      const int rowoff = (b0 + em) * Hh + eh0;
      const float4 cc  = *(const float4*)(cs + p * Bb * Hh + rowoff);
      const float4 ch  = *(const float4*)(hs + p * Bb * Hh + rowoff);
      const int pp     = (p + STK1 - 1) % STK1;
      const float4 ph  = *(const float4*)(hs + pp * Bb * Hh + rowoff);
      const float opf = (float)op;
      const float aab = fabsf(opf);
      const float wnh = (1.f + opf) * aab;
      const float wph = (1.f - opf) * aab;
      const float wch = 1.f - aab;
      float4 nh4, nc4, rt4;
      float* nhp = &nh4.x; float* ncp = &nc4.x; float* rtp = &rt4.x;
      const float* ccp = &cc.x; const float* chp = &ch.x; const float* php = &ph.x;
#pragma unroll
      for (int j = 0; j < 4; ++j) {
        const float ig = lg[em][eh0 + j];
        const float fg = lg[em][Hh + eh0 + j];
        const float gg = lg[em][2 * Hh + eh0 + j];
        const float og = lg[em][3 * Hh + eh0 + j];
        const float nc = sigf(fg) * ccp[j] + sigf(ig) * tanhf(gg);
        const float nh = sigf(og) * tanhf(nc);
        ncp[j] = nc; nhp[j] = nh;
        rtp[j] = nh * wnh + php[j] * wph + chp[j] * wch;
      }
      *(float4*)(hs + (p + 1) * Bb * Hh + rowoff) = nh4;
      *(float4*)(cs + (p + 1) * Bb * Hh + rowoff) = nc4;
      *(float4*)(out + s * Bb * Hh + rowoff)      = rt4;
    }
    __syncthreads();
  }
}

extern "C" void kernel_launch(void* const* d_in, const int* in_sizes, int n_in,
                              void* d_out, int out_size, void* d_ws, size_t ws_size,
                              hipStream_t stream) {
  (void)in_sizes; (void)n_in; (void)out_size; (void)ws_size;
  const float* xall = (const float*)d_in[0];
  const int*   ops  = (const int*)  d_in[1];
  const float* Wih  = (const float*)d_in[2];
  const float* Whh  = (const float*)d_in[3];
  const float* bih  = (const float*)d_in[4];
  const float* bhh  = (const float*)d_in[5];
  const float* ih   = (const float*)d_in[6];
  const float* ic   = (const float*)d_in[7];

  float* hsk = (float*)d_ws;                       // h stack: STK1*B*H f32
  float* csk = hsk + (size_t)STK1 * Bb * Hh;       // c stack
  float* out = (float*)d_out;

  const int n = STK1 * Bb * Hh;
  stack_init<<<(n + 255) / 256, 256, 0, stream>>>(hsk, csk, ih, ic);
  stacklstm<<<Bb / 16, 512, 0, stream>>>(xall, ops, Wih, Whh, bih, bhh, hsk, csk, out);
}
